// CovidRNN_80925773791671
// MI455X (gfx1250) — matrix operand order repaired and checked by reading, not compiled, not run on hardware
//
#include <hip/hip_runtime.h>
#include <hip/hip_bf16.h>
#include <math.h>

// ---------------- problem constants (from reference) ----------------
#define T_   20
#define N_   10000
#define E_   320000
#define XD_  64
#define HD_  256
#define ZD_  256
#define TD_  8
#define YH_  8
#define GD_  768      // 3*HD
#define UK_  288      // ZD+TD+YH=272 padded to multiple of 32
#define PSH_ 100

typedef __bf16 v16bf __attribute__((ext_vector_type(16)));
typedef __bf16 v8bf  __attribute__((ext_vector_type(8)));
typedef float  v8f   __attribute__((ext_vector_type(8)));

// ---------------- WMMA fragment loaders (wave32 layouts per CDNA5 ISA) ----
// A 16x32 bf16: lane(h=lane>>4, r=lane&15) row r, elems[0..7]=K(8h+0..7),
// elems[8..15]=K(16+8h+0..7)  -> two contiguous 16B loads.
__device__ __forceinline__ v16bf load_frag_a(const __bf16* base) {
  v8bf lo = *(const v8bf*)(base);
  v8bf hi = *(const v8bf*)(base + 16);
  v16bf r;
#pragma unroll
  for (int i = 0; i < 8; ++i) { r[i] = lo[i]; r[i + 8] = hi[i]; }
  return r;
}
// B 32x16 bf16 from transposed weights Bt[Nout,K]: lane holds col n0+r,
// elems[0..7]=K(8h+0..7), elems[8..15]=K(16+8h+0..7) -> two 16B loads.
__device__ __forceinline__ v16bf load_frag_b(const __bf16* base) {
  v8bf lo = *(const v8bf*)(base);
  v8bf hi = *(const v8bf*)(base + 16);
  v16bf r;
#pragma unroll
  for (int i = 0; i < 8; ++i) { r[i] = lo[i]; r[i + 8] = hi[i]; }
  return r;
}
__device__ __forceinline__ v8f zero8() {
  v8f z;
#pragma unroll
  for (int g = 0; g < 8; ++g) z[g] = 0.0f;
  return z;
}
__device__ __forceinline__ v8f wmma_bf16(v16bf a, v16bf b, v8f c) {
  return __builtin_amdgcn_wmma_f32_16x16x32_bf16(false, a, false, b, (short)0, c,
                                                 false, false);
}

// ---------------- bf16 WMMA GEMM:  C[M,Nout] = act(A[M,K] * Bt^T + bias) ----
// A row-major bf16 [M,K] (K multiple of 32), Bt row-major bf16 [Nout,K].
// grid = (Nout/64, ceil(M/64)); block = 128 (4 waves). Wave w of a block
// computes the 64(M) x 16(N) strip at n-tile blockIdx.x*4+w, rows blockIdx.y*64.
// All tile indices are wave-uniform (blockIdx + readfirstlane) so WMMA runs
// with full EXEC and scalar branches only.
__global__ void __launch_bounds__(128)
gemm_bf16_wmma(const __bf16* __restrict__ A, const __bf16* __restrict__ Bt,
               const float* __restrict__ bias, float* __restrict__ C,
               int M, int K, int Nout, int do_relu, int has_bias) {
  const int lane   = threadIdx.x & 31;
  const int waveid = __builtin_amdgcn_readfirstlane((int)(threadIdx.x >> 5));
  const int tn = blockIdx.x * 4 + waveid;    // N tile (uniform)
  const int m0 = blockIdx.y * 64;            // M base (uniform)
  const int n0 = tn << 4;
  const int h  = lane >> 4;
  const int r  = lane & 15;

  const __bf16* bptr = Bt + (size_t)(n0 + r) * K + 8 * h;
  const __bf16* aptr = A  + (size_t)(m0 + r) * K + 8 * h;
  const float bv = has_bias ? bias[n0 + r] : 0.0f;
  const size_t arow = (size_t)16 * K;

  if (M - m0 >= 64) {
    // -------- hot path: 4 full 16-row tiles, software-pipelined K loop -----
    v8f c0 = zero8(), c1 = zero8(), c2 = zero8(), c3 = zero8();
    v16bf bcur = load_frag_b(bptr);
    v16bf a0 = load_frag_a(aptr);
    v16bf a1 = load_frag_a(aptr + arow);
    v16bf a2 = load_frag_a(aptr + 2 * arow);
    v16bf a3 = load_frag_a(aptr + 3 * arow);
    for (int kc = 32; kc < K; kc += 32) {
      v16bf bn  = load_frag_b(bptr + kc);
      v16bf a0n = load_frag_a(aptr + kc);
      v16bf a1n = load_frag_a(aptr + arow + kc);
      v16bf a2n = load_frag_a(aptr + 2 * arow + kc);
      v16bf a3n = load_frag_a(aptr + 3 * arow + kc);
      c0 = wmma_bf16(a0, bcur, c0);
      c1 = wmma_bf16(a1, bcur, c1);
      c2 = wmma_bf16(a2, bcur, c2);
      c3 = wmma_bf16(a3, bcur, c3);
      bcur = bn; a0 = a0n; a1 = a1n; a2 = a2n; a3 = a3n;
    }
    c0 = wmma_bf16(a0, bcur, c0);
    c1 = wmma_bf16(a1, bcur, c1);
    c2 = wmma_bf16(a2, bcur, c2);
    c3 = wmma_bf16(a3, bcur, c3);

    v8f cc[4] = {c0, c1, c2, c3};
#pragma unroll
    for (int i = 0; i < 4; ++i) {
      const int rowbase = m0 + i * 16 + 8 * h;  // + g below
#pragma unroll
      for (int g = 0; g < 8; ++g) {
        float v = cc[i][g] + bv;
        if (do_relu) v = fmaxf(v, 0.0f);
        C[(size_t)(rowbase + g) * Nout + (n0 + r)] = v;
      }
    }
  } else {
    // -------- tail: 1..3 full 16-row tiles (M % 16 == 0 in all our calls) --
    const int mvalid = (M - m0) >> 4;          // uniform
    for (int i = 0; i < mvalid; ++i) {
      v8f c = zero8();
      const __bf16* ap = aptr + (size_t)i * arow;
      for (int kc = 0; kc < K; kc += 32)
        c = wmma_bf16(load_frag_a(ap + kc), load_frag_b(bptr + kc), c);
      const int rowbase = m0 + i * 16 + 8 * h;
#pragma unroll
      for (int g = 0; g < 8; ++g) {
        float v = c[g] + bv;
        if (do_relu) v = fmaxf(v, 0.0f);
        C[(size_t)(rowbase + g) * Nout + (n0 + r)] = v;
      }
    }
  }
}

// ---------------- small utility kernels ----------------
__global__ void k_zero(float* p, long n) {
  long i = (long)blockIdx.x * blockDim.x + threadIdx.x;
  if (i < n) p[i] = 0.0f;
}
__global__ void k_copy(const float* __restrict__ s, float* __restrict__ d, long n) {
  long i = (long)blockIdx.x * blockDim.x + threadIdx.x;
  if (i < n) d[i] = s[i];
}
__global__ void k_cast(const float* __restrict__ s, __bf16* __restrict__ d, long n) {
  long i = (long)blockIdx.x * blockDim.x + threadIdx.x;
  if (i < n) d[i] = (__bf16)s[i];
}
// Bt[j,k] = (bf16) W[k,j]   (W is [K,Nout])
__global__ void k_transpose_cast(const float* __restrict__ W, __bf16* __restrict__ Bt,
                                 int K, int Nout) {
  long i = (long)blockIdx.x * blockDim.x + threadIdx.x;
  if (i >= (long)K * Nout) return;
  int k = (int)(i / Nout), j = (int)(i % Nout);
  Bt[(size_t)j * K + k] = (__bf16)W[i];
}
// O[r, 0..Cp) = c<C ? W[r,c] : 0    (row-count R, pad K-dim C -> Cp)
__global__ void k_cast_pad_rows(const float* __restrict__ W, __bf16* __restrict__ O,
                                int R, int C, int Cp) {
  long i = (long)blockIdx.x * blockDim.x + threadIdx.x;
  if (i >= (long)R * Cp) return;
  int rr = (int)(i / Cp), c = (int)(i % Cp);
  O[i] = (c < C) ? (__bf16)W[(size_t)rr * C + c] : (__bf16)0.0f;
}
// dst[N,512] = bf16 concat(h[N,256], rep[N,256])
__global__ void k_concat2(const float* __restrict__ a, const float* __restrict__ b,
                          __bf16* __restrict__ d, int n_rows) {
  long i = (long)blockIdx.x * blockDim.x + threadIdx.x;
  if (i >= (long)n_rows * 512) return;
  int nrow = (int)(i >> 9), c = (int)(i & 511);
  float v = (c < 256) ? a[(size_t)nrow * 256 + c] : b[(size_t)nrow * 256 + (c - 256)];
  d[i] = (__bf16)v;
}
// dst[N,288] = bf16 concat(z[N,256], Ct[N,8], Yt[N,8], zeros[16])
__global__ void k_build_u(const float* __restrict__ z, const float* __restrict__ Ct,
                          const float* __restrict__ Yt, __bf16* __restrict__ d, int n_rows) {
  long i = (long)blockIdx.x * blockDim.x + threadIdx.x;
  if (i >= (long)n_rows * UK_) return;
  int nrow = (int)(i / UK_), c = (int)(i % UK_);
  float v = 0.0f;
  if (c < 256)      v = z[(size_t)nrow * 256 + c];
  else if (c < 264) v = Ct[(size_t)nrow * 8 + (c - 256)];
  else if (c < 272) v = Yt[(size_t)nrow * 8 + (c - 264)];
  d[i] = (__bf16)v;
}
// rep = relu(agg + b[col])
__global__ void k_bias_relu(const float* __restrict__ agg, const float* __restrict__ b,
                            float* __restrict__ rep, long n) {
  long i = (long)blockIdx.x * blockDim.x + threadIdx.x;
  if (i < n) rep[i] = fmaxf(agg[i] + b[i & 255], 0.0f);
}

// ---------------- edge scatter: agg[dst] += val * support[src] ----------------
__global__ void k_edge_scatter(const float* __restrict__ support,
                               const float* __restrict__ vals,
                               const int* __restrict__ src, const int* __restrict__ dst,
                               float* __restrict__ agg, int E) {
  long tid = (long)blockIdx.x * blockDim.x + threadIdx.x;
  int e = (int)(tid >> 6);
  if (e >= E) return;
  int c4 = (int)(tid & 63) << 2;
  int s = src[e], d = dst[e];
  float v = vals[e];
  if (e + 1 < E)  // hint next source row into cache: lowers to global_prefetch_b8
    __builtin_prefetch(support + (size_t)src[e + 1] * 256 + c4, 0, 1);
  const float4 sv = *(const float4*)(support + (size_t)s * 256 + c4);
  float* ap = agg + (size_t)d * 256 + c4;
  atomicAdd(ap + 0, v * sv.x);
  atomicAdd(ap + 1, v * sv.y);
  atomicAdd(ap + 2, v * sv.z);
  atomicAdd(ap + 3, v * sv.w);
}

// ---------------- GRU pointwise update (torch semantics) ----------------
__device__ __forceinline__ float sigm(float x) { return 1.0f / (1.0f + __expf(-x)); }

__global__ void k_gru(const float* __restrict__ gx, const float* __restrict__ gh,
                      float* __restrict__ h, int n_rows) {
  long i = (long)blockIdx.x * blockDim.x + threadIdx.x;
  if (i >= (long)n_rows * 256) return;
  int nrow = (int)(i >> 8), j = (int)(i & 255);
  const float* gxr = gx + (size_t)nrow * GD_;
  const float* ghr = gh + (size_t)nrow * GD_;
  float r  = sigm(gxr[j] + ghr[j]);
  float zt = sigm(gxr[256 + j] + ghr[256 + j]);
  float nn = tanhf(gxr[512 + j] + r * ghr[512 + j]);
  float hv = h[i];
  h[i] = (1.0f - zt) * nn + zt * hv;
}

// ---------------- y0/y1 heads: per-row 256-dot, one wave32 per row ----------
__global__ void k_heads(const float* __restrict__ y00, const float* __restrict__ y10,
                        const float* __restrict__ W01, const float* __restrict__ W11,
                        const float* __restrict__ b01, const float* __restrict__ b11,
                        float* __restrict__ Y0, float* __restrict__ Y1, int n_rows) {
  int wave = (int)(((long)blockIdx.x * blockDim.x + threadIdx.x) >> 5);
  int lane = threadIdx.x & 31;
  if (wave >= n_rows) return;
  float s0 = 0.0f, s1 = 0.0f;
  for (int k = lane; k < 256; k += 32) {
    s0 += y00[(size_t)wave * 256 + k] * W01[k];
    s1 += y10[(size_t)wave * 256 + k] * W11[k];
  }
#pragma unroll
  for (int off = 16; off > 0; off >>= 1) {
    s0 += __shfl_down(s0, off, 32);
    s1 += __shfl_down(s1, off, 32);
  }
  if (lane == 0) { Y0[wave] = s0 + b01[0]; Y1[wave] = s1 + b11[0]; }
}

// ---------------- domain classifier + softmax (one row per block, LDS) -------
__global__ void __launch_bounds__(128)
k_ps(const float* __restrict__ z, const float* __restrict__ W1,
     const float* __restrict__ b1, const float* __restrict__ g,
     const float* __restrict__ bb, const float* __restrict__ W2,
     const float* __restrict__ b2, float* __restrict__ PS, int n_rows) {
  __shared__ float zrow[256];
  __shared__ float sig[PSH_];
  __shared__ float logits[2];
  int nrow = blockIdx.x;
  if (nrow >= n_rows) return;
  for (int c = threadIdx.x; c < 256; c += blockDim.x)
    zrow[c] = z[(size_t)nrow * 256 + c];
  __syncthreads();
  int j = threadIdx.x;
  if (j < PSH_) {
    float acc = b1[j];
    for (int k = 0; k < 256; ++k) acc += zrow[k] * W1[(size_t)k * PSH_ + j];
    float bn = g[j] * rsqrtf(1.0f + 1e-5f) * acc + bb[j];
    sig[j] = sigm(bn);
  }
  __syncthreads();
  if (j < 2) {
    float acc = b2[j];
    for (int k = 0; k < PSH_; ++k) acc += sig[k] * W2[(size_t)k * 2 + j];
    logits[j] = acc;
  }
  __syncthreads();
  if (j == 0) {
    float m  = fmaxf(logits[0], logits[1]);
    float e0 = __expf(logits[0] - m), e1 = __expf(logits[1] - m);
    float inv = 1.0f / (e0 + e1);
    PS[(size_t)nrow * 2 + 0] = e0 * inv;
    PS[(size_t)nrow * 2 + 1] = e1 * inv;
  }
}

// ---------------- host launcher ----------------
static inline int cdiv(long a, long b) { return (int)((a + b - 1) / b); }

extern "C" void kernel_launch(void* const* d_in, const int* in_sizes, int n_in,
                              void* d_out, int out_size, void* d_ws, size_t ws_size,
                              hipStream_t stream) {
  // inputs (setup_inputs dict order; params flattened in insertion order)
  const float* X     = (const float*)d_in[0];
  const float* Cfeat = (const float*)d_in[1];
  const float* Yhist = (const float*)d_in[2];
  const float* ev    = (const float*)d_in[3];
  const int*   es    = (const int*)d_in[4];
  const int*   ed    = (const int*)d_in[5];
  const float* phi_W = (const float*)d_in[6];
  const float* phi_b = (const float*)d_in[7];
  const float* gc_W  = (const float*)d_in[8];
  const float* gc_b  = (const float*)d_in[9];
  const float* fuse_W= (const float*)d_in[10];
  const float* fuse_b= (const float*)d_in[11];
  const float* W00   = (const float*)d_in[12];
  const float* b00   = (const float*)d_in[13];
  const float* W10   = (const float*)d_in[14];
  const float* b10   = (const float*)d_in[15];
  const float* W01   = (const float*)d_in[16];
  const float* b01   = (const float*)d_in[17];
  const float* W11   = (const float*)d_in[18];
  const float* b11   = (const float*)d_in[19];
  const float* psW1  = (const float*)d_in[20];
  const float* psb1  = (const float*)d_in[21];
  const float* bng   = (const float*)d_in[22];
  const float* bnb   = (const float*)d_in[23];
  const float* psW2  = (const float*)d_in[24];
  const float* psb2  = (const float*)d_in[25];
  const float* Wih   = (const float*)d_in[26];
  const float* bih   = (const float*)d_in[27];
  const float* Whh   = (const float*)d_in[28];
  const float* bhh   = (const float*)d_in[29];
  (void)in_sizes; (void)n_in; (void)out_size; (void)ws_size;

  // outputs: (Y1, Y0, Z, PS, h_fin) concatenated flat
  float* out = (float*)d_out;
  float* Y1o = out;
  float* Y0o = out + (size_t)T_ * N_;
  float* Zo  = out + 2ull * T_ * N_;
  float* PSo = Zo  + (size_t)T_ * N_ * ZD_;
  float* Hfo = PSo + 2ull * T_ * N_;

  // workspace carving (256B aligned)
  char* wsb = (char*)d_ws;
  size_t off = 0;
  auto carve = [&](size_t bytes) -> char* {
    char* p = wsb + off;
    off += (bytes + 255) & ~(size_t)255;
    return p;
  };
  // bf16 pre-packed weights (all stored as Bt[Nout,K] row-major)
  __bf16* phiT  = (__bf16*)carve((size_t)HD_ * XD_ * 2);
  __bf16* gcT0  = (__bf16*)carve((size_t)HD_ * HD_ * 2);
  __bf16* gcT1  = (__bf16*)carve((size_t)HD_ * HD_ * 2);
  __bf16* fuseT = (__bf16*)carve((size_t)ZD_ * (2 * HD_) * 2);
  __bf16* w00T  = (__bf16*)carve((size_t)ZD_ * ZD_ * 2);
  __bf16* w10T  = (__bf16*)carve((size_t)ZD_ * ZD_ * 2);
  __bf16* wihP  = (__bf16*)carve((size_t)GD_ * UK_ * 2);  // already [Nout,K]
  __bf16* whhC  = (__bf16*)carve((size_t)GD_ * HD_ * 2);  // already [Nout,K]
  // bf16 activations
  __bf16* A64   = (__bf16*)carve((size_t)N_ * XD_ * 2);
  __bf16* A256  = (__bf16*)carve((size_t)N_ * 256 * 2);
  __bf16* A512  = (__bf16*)carve((size_t)N_ * 512 * 2);
  __bf16* A288  = (__bf16*)carve((size_t)N_ * UK_ * 2);
  // f32 state / intermediates
  float* h    = (float*)carve((size_t)N_ * HD_ * 4);
  float* rep  = (float*)carve((size_t)N_ * HD_ * 4);
  float* sup  = (float*)carve((size_t)N_ * HD_ * 4);
  float* agg  = (float*)carve((size_t)N_ * HD_ * 4);
  float* gx   = (float*)carve((size_t)N_ * GD_ * 4);
  float* gh   = (float*)carve((size_t)N_ * GD_ * 4);
  float* y00  = (float*)carve((size_t)N_ * ZD_ * 4);
  float* y10  = (float*)carve((size_t)N_ * ZD_ * 4);

  const int TB = 256;
  auto gemm = [&](const __bf16* A, const __bf16* Bt, const float* bias, float* Cm,
                  int M, int K, int Nout, int relu, int hasb) {
    dim3 grid(Nout / 64, (unsigned)cdiv(M, 64));  // Nout is 256 or 768 here
    gemm_bf16_wmma<<<grid, 128, 0, stream>>>(A, Bt, bias, Cm, M, K, Nout, relu, hasb);
  };

  // ---- one-time weight prep (deterministic; re-run every call) ----
  k_transpose_cast<<<cdiv((long)XD_ * HD_, TB), TB, 0, stream>>>(phi_W, phiT, XD_, HD_);
  k_transpose_cast<<<cdiv((long)HD_ * HD_, TB), TB, 0, stream>>>(gc_W, gcT0, HD_, HD_);
  k_transpose_cast<<<cdiv((long)HD_ * HD_, TB), TB, 0, stream>>>(gc_W + (size_t)HD_ * HD_, gcT1, HD_, HD_);
  k_transpose_cast<<<cdiv((long)2 * HD_ * ZD_, TB), TB, 0, stream>>>(fuse_W, fuseT, 2 * HD_, ZD_);
  k_transpose_cast<<<cdiv((long)ZD_ * ZD_, TB), TB, 0, stream>>>(W00 + (size_t)ZD_ * ZD_, w00T, ZD_, ZD_);
  k_transpose_cast<<<cdiv((long)ZD_ * ZD_, TB), TB, 0, stream>>>(W10 + (size_t)ZD_ * ZD_, w10T, ZD_, ZD_);
  k_cast_pad_rows<<<cdiv((long)GD_ * UK_, TB), TB, 0, stream>>>(Wih, wihP, GD_, ZD_ + TD_ + YH_, UK_);
  k_cast_pad_rows<<<cdiv((long)GD_ * HD_, TB), TB, 0, stream>>>(Whh, whhC, GD_, HD_, HD_);
  k_zero<<<cdiv((long)N_ * HD_, TB), TB, 0, stream>>>(h, (long)N_ * HD_);

  const __bf16* gcT[2] = {gcT0, gcT1};

  for (int t = 0; t < T_; ++t) {
    const float* Xt = X     + (size_t)t * N_ * XD_;
    const float* Ct = Cfeat + (size_t)t * N_ * TD_;
    const float* Yt = Yhist + (size_t)t * N_ * YH_;
    const float* vt = ev + (size_t)t * E_;
    const int*   st = es + (size_t)t * E_;
    const int*   dt = ed + (size_t)t * E_;
    float* zt = Zo + (size_t)t * N_ * ZD_;

    // phi = relu(X @ phi_W + b)
    k_cast<<<cdiv((long)N_ * XD_, TB), TB, 0, stream>>>(Xt, A64, (long)N_ * XD_);
    gemm(A64, phiT, phi_b, rep, N_, XD_, HD_, 1, 1);

    // 2x graph conv: support = rep @ W; agg = scatter; rep = relu(agg + b)
    for (int l = 0; l < 2; ++l) {
      k_cast<<<cdiv((long)N_ * HD_, TB), TB, 0, stream>>>(rep, A256, (long)N_ * HD_);
      gemm(A256, gcT[l], nullptr, sup, N_, HD_, HD_, 0, 0);
      k_zero<<<cdiv((long)N_ * HD_, TB), TB, 0, stream>>>(agg, (long)N_ * HD_);
      k_edge_scatter<<<cdiv((long)E_ * 64, TB), TB, 0, stream>>>(sup, vt, st, dt, agg, E_);
      k_bias_relu<<<cdiv((long)N_ * HD_, TB), TB, 0, stream>>>(agg, gc_b + (size_t)l * HD_, rep,
                                                               (long)N_ * HD_);
    }

    // z = relu([h, rep] @ fuse_W + b) -> written straight into output Z[t]
    k_concat2<<<cdiv((long)N_ * 512, TB), TB, 0, stream>>>(h, rep, A512, N_);
    gemm(A512, fuseT, fuse_b, zt, N_, 2 * HD_, ZD_, 1, 1);

    // GRU: gx = u @ W_ih^T + b_ih ; gh = h @ W_hh^T + b_hh ; pointwise update
    k_build_u<<<cdiv((long)N_ * UK_, TB), TB, 0, stream>>>(zt, Ct, Yt, A288, N_);
    gemm(A288, wihP, bih, gx, N_, UK_, GD_, 0, 1);
    k_cast<<<cdiv((long)N_ * HD_, TB), TB, 0, stream>>>(h, A256, (long)N_ * HD_);
    gemm(A256, whhC, bhh, gh, N_, HD_, GD_, 0, 1);
    k_gru<<<cdiv((long)N_ * HD_, TB), TB, 0, stream>>>(gx, gh, h, N_);

    // heads (last n_out index only reaches the output)
    k_cast<<<cdiv((long)N_ * ZD_, TB), TB, 0, stream>>>(zt, A256, (long)N_ * ZD_);
    gemm(A256, w00T, b00 + ZD_, y00, N_, ZD_, ZD_, 1, 1);
    gemm(A256, w10T, b10 + ZD_, y10, N_, ZD_, ZD_, 1, 1);
    k_heads<<<cdiv((long)N_ * 32, 128), 128, 0, stream>>>(y00, y10, W01, W11, b01, b11,
                                                          Y0o + (size_t)t * N_,
                                                          Y1o + (size_t)t * N_, N_);

    // domain classifier + softmax
    k_ps<<<N_, 128, 0, stream>>>(zt, psW1, psb1, bng, bnb, psW2, psb2,
                                 PSo + (size_t)t * N_ * 2, N_);
  }

  // h_fin
  k_copy<<<cdiv((long)N_ * HD_, TB), TB, 0, stream>>>(h, Hfo, (long)N_ * HD_);
}
